// LocalFacialExtractor_19559281066095
// MI455X (gfx1250) — matrix operand order repaired and checked by reading, not compiled, not run on hardware
//
#include <hip/hip_runtime.h>
#include <cstdint>

// ---------------------------------------------------------------------------
// CDNA5 (gfx1250) implementation of LocalFacialExtractor forward.
// All dense math goes through V_WMMA_F32_16X16X32_BF16 (wave32).
// ---------------------------------------------------------------------------

typedef __attribute__((ext_vector_type(16))) __bf16 v16bf;
typedef __attribute__((ext_vector_type(8)))  float  v8f;
typedef __attribute__((ext_vector_type(2)))  float  f32x2;
typedef __attribute__((ext_vector_type(2)))  __bf16 bf16x2;

__device__ __forceinline__ bf16x2 cvt2(float x, float y) {
  f32x2 t; t[0] = x; t[1] = y;
  return __builtin_convertvector(t, bf16x2);   // -> v_cvt_pk_bf16_f32
}

#define BM 128
#define BN 128
#define BK 32
#define LPAD 8
#define ASTRIDE (BK + LPAD)   // 40 bf16 per row -> 80B, 16B aligned
#define BSTRIDE (BK + LPAD)

// ------------------------------- GEMM --------------------------------------
// C[M,N] = A[M,K] @ B[K,N] (+ bias[N]) (+ R[M,N]).  f32 in, bf16 WMMA, f32 out.
// Block: 256 threads (8 waves), tile 128x128, K-step 32.
// Wave (wm,wn) owns rows wm*64..+63, cols wn*32..+31 -> 4x2 wmma tiles.
template <bool HAS_BIAS, bool HAS_RES>
__global__ __launch_bounds__(256) void gemm_bf16_kernel(
    const float* __restrict__ A, const float* __restrict__ B,
    const float* __restrict__ bias, const float* __restrict__ R,
    float* __restrict__ C, int M, int N, int K, int lda, int ldb, int ldc) {
  __shared__ __align__(16) __bf16 As[BM * ASTRIDE];
  __shared__ __align__(16) __bf16 Bs[BN * BSTRIDE];

  const int tid  = threadIdx.x;
  const int lane = tid & 31;
  const int wave = tid >> 5;
  const int wm   = wave >> 2;   // 0..1
  const int wn   = wave & 3;    // 0..3
  const int l16  = lane & 15;
  const int lhalf = lane >> 4;  // 0 or 1
  const int bm = blockIdx.y * BM;
  const int bn = blockIdx.x * BN;

  // A staging: thread -> (row = tid/2, 16 contiguous k)
  const int a_row = tid >> 1;
  const int a_kk  = (tid & 1) * 16;
  const bool a_in = (bm + a_row) < M;
  // B staging: thread -> 4(k) x 4(n) micro-tile, transposed into Bs[n][k]
  const int b_k0 = (tid >> 5) * 4;   // 0,4,..,28
  const int b_n0 = (tid & 31) * 4;   // 0..124

  // Zero-fill OOB A rows once; never rewritten inside the K loop.
  if (!a_in) {
    const uint4 z = make_uint4(0, 0, 0, 0);
    uint4* dst = (uint4*)&As[a_row * ASTRIDE + a_kk];
    dst[0] = z;
    dst[1] = z;
  }

  const float* ap_base = A + (long)(bm + a_row) * lda + a_kk;
  const float* bp_base = B + (long)b_k0 * ldb + bn + b_n0;

  v8f acc[4][2] = {};

  for (int k0 = 0; k0 < K; k0 += BK) {
    // ---- stage A tile (BM x BK): 4x b128 load, pk-convert, 2x ds_store_b128 -
    if (a_in) {
      const float4* ap = (const float4*)(ap_base + k0);
      const float4 f0 = ap[0], f1 = ap[1], f2 = ap[2], f3 = ap[3];
      union { bf16x2 b[8]; uint4 u[2]; } pk;
      pk.b[0] = cvt2(f0.x, f0.y); pk.b[1] = cvt2(f0.z, f0.w);
      pk.b[2] = cvt2(f1.x, f1.y); pk.b[3] = cvt2(f1.z, f1.w);
      pk.b[4] = cvt2(f2.x, f2.y); pk.b[5] = cvt2(f2.z, f2.w);
      pk.b[6] = cvt2(f3.x, f3.y); pk.b[7] = cvt2(f3.z, f3.w);
      uint4* dst = (uint4*)&As[a_row * ASTRIDE + a_kk];
      dst[0] = pk.u[0];
      dst[1] = pk.u[1];
      if (k0 + BK < K) __builtin_prefetch(ap_base + k0 + BK, 0, 1);
    }
    // ---- stage B tile (BK x BN) -> Bs[n][k]: 4x b128, transpose, 4x b64 ----
    {
      float v[4][4];  // [k][n]
#pragma unroll
      for (int kk = 0; kk < 4; ++kk) {
        const float4 f = *(const float4*)(bp_base + (long)(k0 + kk) * ldb);
        v[kk][0] = f.x; v[kk][1] = f.y; v[kk][2] = f.z; v[kk][3] = f.w;
      }
#pragma unroll
      for (int nn = 0; nn < 4; ++nn) {
        union { bf16x2 b[2]; uint2 u; } pk;
        pk.b[0] = cvt2(v[0][nn], v[1][nn]);
        pk.b[1] = cvt2(v[2][nn], v[3][nn]);
        *(uint2*)&Bs[(b_n0 + nn) * BSTRIDE + b_k0] = pk.u;
      }
      if (k0 + BK < K) __builtin_prefetch(bp_base + (long)(k0 + BK) * ldb, 0, 1);
    }
    __syncthreads();

    // ---- fragments per ISA 16-bit layouts ----
    // A 16x32: lane<16 -> K 0..7 & 16..23 ; lane>=16 -> K 8..15 & 24..31
    v16bf afrag[4];
#pragma unroll
    for (int t = 0; t < 4; ++t) {
      const int row = wm * 64 + t * 16 + l16;
      const int kb = lhalf * 8;
      union { v16bf v; uint4 u[2]; } tmp;
      tmp.u[0] = *(const uint4*)&As[row * ASTRIDE + kb];
      tmp.u[1] = *(const uint4*)&As[row * ASTRIDE + kb + 16];
      afrag[t] = tmp.v;
    }
    // B 32x16: lane<16 -> K 0..7 & 16..23 of col l16 ; lane>=16 -> K 8..15 & 24..31
    v16bf bfrag[2];
#pragma unroll
    for (int t = 0; t < 2; ++t) {
      const int col = wn * 32 + t * 16 + l16;
      const int kb = lhalf * 8;
      const uint4* p = (const uint4*)&Bs[col * BSTRIDE + kb];
      union { v16bf v; uint4 u[2]; } tmp;
      tmp.u[0] = p[0];
      tmp.u[1] = p[2];
      bfrag[t] = tmp.v;
    }
#pragma unroll
    for (int i = 0; i < 4; ++i)
#pragma unroll
      for (int j = 0; j < 2; ++j)
        acc[i][j] = __builtin_amdgcn_wmma_f32_16x16x32_bf16(
            false, afrag[i], false, bfrag[j], (short)0, acc[i][j], false, false);
    __syncthreads();
  }

  // ---- epilogue: f32 C layout: VGPR r -> M = r + (lane>=16 ? 8:0), N = l16 --
#pragma unroll
  for (int i = 0; i < 4; ++i) {
#pragma unroll
    for (int j = 0; j < 2; ++j) {
      const int col = bn + wn * 32 + j * 16 + l16;
#pragma unroll
      for (int r = 0; r < 8; ++r) {
        const int row = bm + wm * 64 + i * 16 + r + lhalf * 8;
        if (row < M) {
          float v = acc[i][j][r];
          if (HAS_BIAS) v += bias[col];
          if (HAS_RES)  v += R[(long)row * ldc + col];
          C[(long)row * ldc + col] = v;
        }
      }
    }
  }
}

// ----------------------------- LayerNorm -----------------------------------
// One block per 1024-wide row. Output row remapped so LN results can be
// written straight into the concatenated [B,619,1024] kv-context buffer:
//   dst_row = (src_row / src_rpb) * dst_rpb + dst_off + (src_row % src_rpb)
__global__ __launch_bounds__(256) void layernorm_kernel(
    const float* __restrict__ x, const float* __restrict__ g,
    const float* __restrict__ bb, float* __restrict__ y,
    int src_rpb, int dst_rpb, int dst_off, int lrelu) {
  const int row = blockIdx.x;
  const int tid = threadIdx.x;
  const int batch = row / src_rpb;
  const int r = row - batch * src_rpb;
  const float* xs = x + (long)row * 1024;
  float* ys = y + ((long)batch * dst_rpb + dst_off + r) * 1024;

  __shared__ float red[8];
  float v[4];
  float s = 0.f;
#pragma unroll
  for (int u = 0; u < 4; ++u) { v[u] = xs[tid + u * 256]; s += v[u]; }
#pragma unroll
  for (int off = 16; off; off >>= 1) s += __shfl_xor(s, off, 32);
  if ((tid & 31) == 0) red[tid >> 5] = s;
  __syncthreads();
  if (tid == 0) { float t = 0; for (int i = 0; i < 8; ++i) t += red[i]; red[0] = t; }
  __syncthreads();
  const float mean = red[0] * (1.0f / 1024.0f);
  __syncthreads();

  float vs = 0.f;
#pragma unroll
  for (int u = 0; u < 4; ++u) { float d = v[u] - mean; vs += d * d; }
#pragma unroll
  for (int off = 16; off; off >>= 1) vs += __shfl_xor(vs, off, 32);
  if ((tid & 31) == 0) red[tid >> 5] = vs;
  __syncthreads();
  if (tid == 0) { float t = 0; for (int i = 0; i < 8; ++i) t += red[i]; red[0] = t; }
  __syncthreads();
  const float inv = rsqrtf(red[0] * (1.0f / 1024.0f) + 1e-5f);

#pragma unroll
  for (int u = 0; u < 4; ++u) {
    const int c = tid + u * 256;
    float o = (v[u] - mean) * inv * g[c] + bb[c];
    if (lrelu) o = (o > 0.f) ? o : 0.01f * o;
    ys[c] = o;
  }
}

// ----------------------------- Attention -----------------------------------
// Q:[32,37,1024] K,V:[32,619,1024] (16 heads x 64). One block per (b,h),
// one wave per query row; scores staged in LDS; stable softmax; VALU PV.
#define NKEY 619
#define NQROW 37
__global__ __launch_bounds__(256) void attn_kernel(
    const float* __restrict__ Q, const float* __restrict__ Kb,
    const float* __restrict__ Vb, float* __restrict__ O) {
  const int b = blockIdx.x;
  const int h = blockIdx.y;
  const int wave = threadIdx.x >> 5;
  const int lane = threadIdx.x & 31;

  __shared__ float qs[NQROW * 64];   // 9.25 KB
  __shared__ float ps[8][640];       // 20 KB
  for (int idx = threadIdx.x; idx < NQROW * 64; idx += 256) {
    const int qr = idx >> 6, d = idx & 63;
    qs[idx] = Q[((long)b * NQROW + qr) * 1024 + h * 64 + d];
  }
  __syncthreads();

  const float scale = 0.125f;  // (DH^-0.25)^2 = 64^-0.5
  for (int qr = wave; qr < NQROW; qr += 8) {
    float mymax = -1e30f;
    for (int kk = lane; kk < NKEY; kk += 32) {
      const float* kp = Kb + ((long)b * NKEY + kk) * 1024 + h * 64;
      float dot = 0.f;
#pragma unroll 8
      for (int d = 0; d < 64; ++d) dot += qs[qr * 64 + d] * kp[d];
      dot *= scale;
      ps[wave][kk] = dot;
      mymax = fmaxf(mymax, dot);
    }
#pragma unroll
    for (int off = 16; off; off >>= 1) mymax = fmaxf(mymax, __shfl_xor(mymax, off, 32));
    float mysum = 0.f;
    for (int kk = lane; kk < NKEY; kk += 32) {
      const float e = __expf(ps[wave][kk] - mymax);
      ps[wave][kk] = e;
      mysum += e;
    }
#pragma unroll
    for (int off = 16; off; off >>= 1) mysum += __shfl_xor(mysum, off, 32);
    const float rinv = 1.f / mysum;
#pragma unroll
    for (int dd = 0; dd < 2; ++dd) {
      const int d = lane + dd * 32;
      float o = 0.f;
      for (int kk = 0; kk < NKEY; ++kk)
        o += ps[wave][kk] * Vb[((long)b * NKEY + kk) * 1024 + h * 64 + d];
      O[((long)b * NQROW + qr) * 1024 + h * 64 + d] = o * rinv;
    }
  }
}

// ---------------------------- elementwise ----------------------------------
__global__ void gelu_kernel(float* __restrict__ x, long n) {
  const long idx = (long)blockIdx.x * 256 + threadIdx.x;
  if (idx < n) {
    const float v = x[idx];
    x[idx] = 0.5f * v * (1.0f + erff(v * 0.7071067811865476f));
  }
}

__global__ void lat_init_kernel(const float* __restrict__ latents,
                                const float* __restrict__ ide,
                                float* __restrict__ lat) {
  const int idx = blockIdx.x * 256 + threadIdx.x;
  if (idx >= 32 * 37 * 1024) return;
  const int c = idx & 1023;
  const int t = idx >> 10;     // b*37 + r
  const int b = t / 37;
  const int r = t - b * 37;
  lat[idx] = (r < 32) ? latents[r * 1024 + c]
                      : ide[(long)b * 5120 + (r - 32) * 1024 + c];
}

__global__ void slice_kernel(const float* __restrict__ lat, float* __restrict__ out) {
  const int idx = blockIdx.x * 256 + threadIdx.x;
  if (idx >= 32 * 32 * 1024) return;
  const int c = idx & 1023;
  const int t = idx >> 10;     // b*32 + r
  const int b = t >> 5;
  const int r = t & 31;
  out[idx] = lat[((long)b * 37 + r) * 1024 + c];
}

// ---------------------------- orchestration --------------------------------
// Input flattening assumed depth-first in setup_inputs() insertion order:
//  0 id_embeds, 1 vit_hidden_states, 2 latents, 3 proj_out,
//  4 + j*11 : layers[j]{n1g,n1b,n2g,n2b,wq,wkv,wo,ffg,ffb,fw1,fw2}
//  114 + i*10 : mappings[i]{w1,b1,ln1_g,ln1_b,w2,b2,ln2_g,ln2_b,w3,b3}
//  164 : id_map (same 10 keys)

static const size_t T_MAP = 18464ull * 1024;  // mapping activations
static const size_t T_CTX = 32ull * 619 * 1024;
static const size_t T_LAT = 32ull * 37 * 1024;
static const size_t T_FFH = 1184ull * 4096;

extern "C" void kernel_launch(void* const* d_in, const int* in_sizes, int n_in,
                              void* d_out, int out_size, void* d_ws, size_t ws_size,
                              hipStream_t stream) {
  (void)in_sizes; (void)n_in; (void)out_size; (void)ws_size;

  auto in = [&](int i) { return (const float*)d_in[i]; };

  float* base = (float*)d_ws;
  float* tA    = base;                   // [18464,1024]
  float* tB    = tA + T_MAP;             // [18464,1024]
  float* ctx   = tB + T_MAP;             // [32,619,1024] LN'd kv input
  float* kbuf  = ctx + T_CTX;            // [32,619,1024]
  float* vbuf  = kbuf + T_CTX;           // [32,619,1024]
  float* lat   = vbuf + T_CTX;           // [32,37,1024]
  float* latn  = lat + T_LAT;            // [32,37,1024]
  float* qbuf  = latn + T_LAT;           // [32,37,1024]
  float* ao    = qbuf + T_LAT;           // [32,37,1024]
  float* ffh   = ao + T_LAT;             // [1184,4096]
  float* id1   = ffh + T_FFH;            // [32,1024]
  float* id2   = id1 + 32 * 1024;        // [32,1024]
  float* ide   = id2 + 32 * 1024;        // [32,5120]
  float* slice = ide + 32 * 5120;        // [1024,1024]

  auto gemm = [&](const float* A, const float* B, const float* bias,
                  const float* R, float* C, int M, int N, int K,
                  int lda, int ldb, int ldc) {
    dim3 grid(N / 128, (M + 127) / 128), block(256);
    if (bias && R)
      gemm_bf16_kernel<true, true><<<grid, block, 0, stream>>>(A, B, bias, R, C, M, N, K, lda, ldb, ldc);
    else if (bias)
      gemm_bf16_kernel<true, false><<<grid, block, 0, stream>>>(A, B, bias, R, C, M, N, K, lda, ldb, ldc);
    else if (R)
      gemm_bf16_kernel<false, true><<<grid, block, 0, stream>>>(A, B, bias, R, C, M, N, K, lda, ldb, ldc);
    else
      gemm_bf16_kernel<false, false><<<grid, block, 0, stream>>>(A, B, bias, R, C, M, N, K, lda, ldb, ldc);
  };
  auto ln = [&](const float* x, const float* g, const float* b, float* y,
                int rows, int srpb, int drpb, int doff, int lrelu) {
    layernorm_kernel<<<rows, 256, 0, stream>>>(x, g, b, y, srpb, drpb, doff, lrelu);
  };

  const float* id_embeds = in(0);
  const float* vit       = in(1);
  const float* latents   = in(2);
  const float* proj_out  = in(3);
  const int IMB = 164;

  // ---- id path: mlp3(1280 -> 1024 -> 1024 -> 5120) ----
  gemm(id_embeds, in(IMB + 0), in(IMB + 1), nullptr, id1, 32, 1024, 1280, 1280, 1024, 1024);
  ln(id1, in(IMB + 2), in(IMB + 3), id1, 32, 1, 1, 0, 1);
  gemm(id1, in(IMB + 4), in(IMB + 5), nullptr, id2, 32, 1024, 1024, 1024, 1024, 1024);
  ln(id2, in(IMB + 6), in(IMB + 7), id2, 32, 1, 1, 0, 1);
  gemm(id2, in(IMB + 8), in(IMB + 9), nullptr, ide, 32, 5120, 1024, 1024, 5120, 5120);

  // ---- latents: [B,32]=latents tile, [B,32:37]=ide ----
  lat_init_kernel<<<(int)((T_LAT + 255) / 256), 256, 0, stream>>>(latents, ide, lat);

  for (int i = 0; i < 5; ++i) {
    const float* vi = vit + (size_t)i * 32 * 577 * 1024;
    const int MB = 114 + i * 10;
    // mapping mlp3 on [18464,1024]
    gemm(vi, in(MB + 0), in(MB + 1), nullptr, tA, 18464, 1024, 1024, 1024, 1024, 1024);
    ln(tA, in(MB + 2), in(MB + 3), tA, 18464, 1, 1, 0, 1);
    gemm(tA, in(MB + 4), in(MB + 5), nullptr, tB, 18464, 1024, 1024, 1024, 1024, 1024);
    ln(tB, in(MB + 6), in(MB + 7), tB, 18464, 1, 1, 0, 1);
    gemm(tB, in(MB + 8), in(MB + 9), nullptr, tA, 18464, 1024, 1024, 1024, 1024, 1024);  // vf

    for (int j = i * 2; j < i * 2 + 2; ++j) {
      const int LB = 4 + j * 11;
      const float *n1g = in(LB + 0), *n1b = in(LB + 1);
      const float *n2g = in(LB + 2), *n2b = in(LB + 3);
      const float *wq = in(LB + 4), *wkv = in(LB + 5), *wo = in(LB + 6);
      const float *ffg = in(LB + 7), *ffb = in(LB + 8);
      const float *fw1 = in(LB + 9), *fw2 = in(LB + 10);

      // kv input = LN(concat(ide, vf, lat)) written straight into ctx[B,619,1024]
      ln(ide, n1g, n1b, ctx, 160, 5, 619, 0, 0);         // rows 0..4
      ln(tA, n1g, n1b, ctx, 18464, 577, 619, 5, 0);      // rows 5..581
      ln(lat, n2g, n2b, ctx, 1184, 37, 619, 582, 0);     // rows 582..618
      ln(lat, n2g, n2b, latn, 1184, 37, 37, 0, 0);       // for q

      gemm(latn, wq, nullptr, nullptr, qbuf, 1184, 1024, 1024, 1024, 1024, 1024);
      gemm(ctx, wkv, nullptr, nullptr, kbuf, 19808, 1024, 1024, 1024, 2048, 1024);
      gemm(ctx, wkv + 1024, nullptr, nullptr, vbuf, 19808, 1024, 1024, 1024, 2048, 1024);

      attn_kernel<<<dim3(32, 16), 256, 0, stream>>>(qbuf, kbuf, vbuf, ao);
      gemm(ao, wo, nullptr, lat, lat, 1184, 1024, 1024, 1024, 1024, 1024);  // + residual

      ln(lat, ffg, ffb, latn, 1184, 1, 1, 0, 0);
      gemm(latn, fw1, nullptr, nullptr, ffh, 1184, 4096, 1024, 1024, 4096, 4096);
      gelu_kernel<<<(int)((T_FFH + 255) / 256), 256, 0, stream>>>(ffh, (long)T_FFH);
      gemm(ffh, fw2, nullptr, lat, lat, 1184, 1024, 4096, 4096, 1024, 1024);  // + residual
    }
  }

  // ---- output projection: lat[:, :32] @ proj_out -> [32,32,2048] ----
  slice_kernel<<<(32 * 32 * 1024 + 255) / 256, 256, 0, stream>>>(lat, slice);
  gemm(slice, proj_out, nullptr, nullptr, (float*)d_out, 1024, 2048, 1024, 1024, 2048, 2048);
}
